// SelfAttnLayer_64476049048260
// MI455X (gfx1250) — matrix operand order repaired and checked by reading, not compiled, not run on hardware
//
#include <hip/hip_runtime.h>
#include <stdint.h>

// ---------------------------------------------------------------------------
// Self-attention layer for MI455X (gfx1250, wave32, WMMA, async global->LDS).
//   qkv   = x @ Wqkv + b          (WMMA f16 -> f32; V written d-major/transposed)
//   attn  = softmax(q k^T) v      (flash attention, double-buffered async LDS tiles)
//   out   = attn @ (Wproj @ Wfc) + (bproj.Wfc + bfc)   (folded tail)
// ---------------------------------------------------------------------------

typedef __attribute__((ext_vector_type(16))) _Float16 v16h;
typedef __attribute__((ext_vector_type(8)))  _Float16 v8h;
typedef __attribute__((ext_vector_type(8)))  float    v8f;
typedef __attribute__((ext_vector_type(4)))  float    v4f;

#define WMMA_F32_F16(a, b, c) \
  __builtin_amdgcn_wmma_f32_16x16x32_f16(false, (a), false, (b), (short)0, (c), false, false)

// Async copy 16B global -> LDS. GVS mode: mem = SGPR base + VGPR off + ioff;
// LDS dest = VGPR[vdst] + ioff. Tracked by ASYNCcnt.
__device__ __forceinline__ void async_b128(uint32_t lds_addr, uint32_t voff,
                                           const void* sbase, int ioff) {
  asm volatile("global_load_async_to_lds_b128 %0, %1, %2 offset:%3"
               :: "v"(lds_addr), "v"(voff), "s"(sbase), "n"(ioff) : "memory");
}
__device__ __forceinline__ void wait_async0() {
  asm volatile("s_wait_asynccnt 0x0" ::: "memory");
}
__device__ __forceinline__ uint32_t lds_off(const void* p) {  // low 32 bits = LDS byte addr
  return (uint32_t)(uintptr_t)p;
}

// B fragment: 16 contiguous halves (K = 16*g + e for half e), contiguous k order (A must use the same).
__device__ __forceinline__ v16h ld_contig16(const _Float16* p) {
  v8h lo = *(const v8h*)p;
  v8h hi = *(const v8h*)(p + 8);
  v16h r;
#pragma unroll
  for (int i = 0; i < 8; i++) { r[i] = lo[i]; r[i + 8] = hi[i]; }
  return r;
}

// A fragment: halves map to K = 8g + e (e<8) and K = 16 + 8g + (e-8) (e>=8).
__device__ __forceinline__ v16h ld_afrag(const _Float16* rowp, int g) {
  v8h lo = *(const v8h*)(rowp + 8 * g);
  v8h hi = *(const v8h*)(rowp + 16 + 8 * g);
  v16h r;
#pragma unroll
  for (int i = 0; i < 8; i++) { r[i] = lo[i]; r[i + 8] = hi[i]; }
  return r;
}

// ---------------------------------------------------------------------------
// Kernel 1: qkv = x @ Wqkv + bqkv.  q,k -> qk[row][512]; v -> vt[b][d][4096].
// grid (128 row-tiles, 8 col-tiles), 256 threads (8 waves of 16 rows x 96 cols)
// ---------------------------------------------------------------------------
__global__ __launch_bounds__(256) void qkv_kernel(const float* __restrict__ x,
                                                  const float* __restrict__ W,
                                                  const float* __restrict__ bias,
                                                  _Float16* __restrict__ qk,
                                                  _Float16* __restrict__ vt) {
  __shared__ _Float16 Wt[96][264];  // [col][d], padded stride 528B (16B-aligned rows)
  const int t  = threadIdx.x;
  const int c0 = blockIdx.y * 96;

  for (int idx = t; idx < 96 * 256; idx += 256) {
    int d = idx / 96, col = idx - d * 96;
    Wt[col][d] = (_Float16)W[(size_t)d * 768 + c0 + col];
  }
  __syncthreads();

  const int wave = t >> 5, lane = t & 31, g = lane >> 4, hl = lane & 15;
  const int grow = blockIdx.x * 128 + wave * 16 + hl;  // A-matrix row for this lane

  v8f acc[6];
#pragma unroll
  for (int cb = 0; cb < 6; cb++) {
    float bv = bias[c0 + cb * 16 + hl];
#pragma unroll
    for (int r = 0; r < 8; r++) acc[cb][r] = bv;
  }

  const float* xr = x + (size_t)grow * 256;
#pragma unroll
  for (int c = 0; c < 8; c++) {  // d chunks of 32
    v4f f0 = *(const v4f*)(xr + 32 * c + 8 * g);
    v4f f1 = *(const v4f*)(xr + 32 * c + 8 * g + 4);
    v4f f2 = *(const v4f*)(xr + 32 * c + 16 + 8 * g);
    v4f f3 = *(const v4f*)(xr + 32 * c + 16 + 8 * g + 4);
    v16h a;
#pragma unroll
    for (int i = 0; i < 4; i++) {
      a[i] = (_Float16)f0[i];      a[4 + i]  = (_Float16)f1[i];
      a[8 + i] = (_Float16)f2[i];  a[12 + i] = (_Float16)f3[i];
    }
#pragma unroll
    for (int cb = 0; cb < 6; cb++) {
      v16h b = ld_afrag(&Wt[cb * 16 + hl][32 * c], g);
      acc[cb] = WMMA_F32_F16(a, b, acc[cb]);
    }
  }

  const int rowbase = blockIdx.x * 128 + wave * 16 + 8 * g;  // D layout row = 8g + r
#pragma unroll
  for (int cb = 0; cb < 6; cb++) {
    const int col = c0 + cb * 16 + hl;  // uniform q/k-vs-v split per cb (16 | 512)
    if (col < 512) {
#pragma unroll
      for (int r = 0; r < 8; r++)
        qk[(size_t)(rowbase + r) * 512 + col] = (_Float16)acc[cb][r];
    } else {
      // transposed V: vt[(batch*256 + d)*4096 + srow]; consecutive r -> contiguous
      _Float16* vp = vt + ((size_t)((rowbase >> 12) * 256 + (col - 512))) * 4096 +
                     (rowbase & 4095);
#pragma unroll
      for (int r = 0; r < 8; r++) vp[r] = (_Float16)acc[cb][r];
    }
  }
}

// ---------------------------------------------------------------------------
// Kernel 2: flash attention. 256 blocks x 128 threads (4 waves).
// Double-buffered K/V tiles staged by async global->LDS B128 copies.
// ---------------------------------------------------------------------------
__global__ __launch_bounds__(128) void attn_kernel(const _Float16* __restrict__ qk,
                                                   const _Float16* __restrict__ vt,
                                                   float* __restrict__ Y) {
  constexpr int S = 4096;
  __shared__ _Float16 Kt[2][32][264];   // [buf][key][d]   row 528B: 16B aligned
  __shared__ _Float16 Vt[2][256][40];   // [buf][d][key]   row 80B: 16B aligned
  __shared__ _Float16 Ps[4][16][40];    // per-wave P scratch [row][key]

  const int t = threadIdx.x;
  const int wave = t >> 5, lane = t & 31, g = lane >> 4, hl = lane & 15;
  const int batch = blockIdx.x >> 6;                     // 64 blocks per batch
  const int qrow0 = (blockIdx.x & 63) * 64 + wave * 16;  // query tile start in S

  // ---- async staging addresses (precomputed; kt advances the global offset) ----
  const int ldr  = t >> 2;        // K loader: key row 0..31
  const int lds0 = (t & 3) * 64;  // K loader: d segment (halves)
  uint32_t k_lds[2], v_lds[2][4], v_voff0[4];
  const uint32_t k_voff0 = (uint32_t)(((batch * S + ldr) * 512 + 256 + lds0) * 2);
#pragma unroll
  for (int p = 0; p < 2; p++) k_lds[p] = lds_off(&Kt[p][ldr][lds0]);
#pragma unroll
  for (int i = 0; i < 4; i++) {
    int chunk = t + 128 * i;            // 512 chunks of 16B cover 256 d-rows x keys 0-7 and 16-23
    int dd = chunk >> 1, seg = (chunk & 1) * 16;
#pragma unroll
    for (int p = 0; p < 2; p++) v_lds[p][i] = lds_off(&Vt[p][dd][seg]);
    v_voff0[i] = (uint32_t)(((batch * 256 + dd) * 4096 + seg) * 2);
  }
  auto issue_tile = [&](int kt, int p) {
    uint32_t kv = k_voff0 + (uint32_t)kt * 1024;  // kt rows * 512 halves * 2B
#pragma unroll
    for (int i = 0; i < 8; i++) async_b128(k_lds[p], kv, qk, 16 * i);
#pragma unroll
    for (int i = 0; i < 4; i++) async_b128(v_lds[p][i], v_voff0[i] + (uint32_t)kt * 2, vt, 0);
  };

  // ---- Q tile 16x256 resident in registers as 8 A-fragments ----
  const _Float16* qrow = qk + ((size_t)(batch * S + qrow0 + hl)) * 512;
  v16h qa[8];
#pragma unroll
  for (int c = 0; c < 8; c++) qa[c] = ld_afrag(qrow + 32 * c, g);

  v8f o[16];
#pragma unroll
  for (int cb = 0; cb < 16; cb++)
#pragma unroll
    for (int r = 0; r < 8; r++) o[cb][r] = 0.f;
  float mi[8], li[8];  // per-row running max / sum (row = 8g + r)
#pragma unroll
  for (int r = 0; r < 8; r++) { mi[r] = -3.0e38f; li[r] = 0.f; }

  issue_tile(0, 0);
  wait_async0();
  __syncthreads();

  int p = 0;
  for (int kt = 0; kt < S; kt += 32) {
    if (kt + 32 < S) issue_tile(kt + 32, p ^ 1);  // prefetch next tile (other buffer)

    // scores S = Q K^T for 2 key sub-blocks of 16
    v8f s0, s1;
#pragma unroll
    for (int r = 0; r < 8; r++) { s0[r] = 0.f; s1[r] = 0.f; }
#pragma unroll
    for (int c = 0; c < 8; c++) {
      v16h b0 = ld_afrag(&Kt[p][hl][32 * c], g);
      v16h b1 = ld_afrag(&Kt[p][16 + hl][32 * c], g);
      s0 = WMMA_F32_F16(qa[c], b0, s0);
      s1 = WMMA_F32_F16(qa[c], b1, s1);
    }

    // online softmax; keys live across lanes within each 16-lane half
#pragma unroll
    for (int r = 0; r < 8; r++) {
      float mx = fmaxf(s0[r], s1[r]);
#pragma unroll
      for (int m = 8; m >= 1; m >>= 1) mx = fmaxf(mx, __shfl_xor(mx, m, 32));
      float mnew = fmaxf(mi[r], mx);
      float corr = __expf(mi[r] - mnew);  // 0 on first tile
      float p0 = __expf(s0[r] - mnew);
      float p1 = __expf(s1[r] - mnew);
      float rs = p0 + p1;
#pragma unroll
      for (int m = 8; m >= 1; m >>= 1) rs += __shfl_xor(rs, m, 32);
      li[r] = li[r] * corr + rs;
      mi[r] = mnew;
#pragma unroll
      for (int cb = 0; cb < 16; cb++) o[cb][r] *= corr;
      Ps[wave][8 * g + r][hl]      = (_Float16)p0;
      Ps[wave][8 * g + r][16 + hl] = (_Float16)p1;
    }

    // O += P V   (wave-private LDS scratch; same-wave LDS is in-order)
    v16h pa = ld_afrag(&Ps[wave][hl][0], g);
#pragma unroll
    for (int cb = 0; cb < 16; cb++) {
      v16h vb = ld_afrag(&Vt[p][cb * 16 + hl][0], g);
      o[cb] = WMMA_F32_F16(pa, vb, o[cb]);
    }

    wait_async0();     // this wave's prefetch landed
    __syncthreads();   // all waves done reading buf p and landing buf p^1
    p ^= 1;
  }

  // normalize and write y (f32)
#pragma unroll
  for (int r = 0; r < 8; r++) {
    float inv = 1.f / li[r];
    const size_t row = (size_t)(batch * S + qrow0 + 8 * g + r);
#pragma unroll
    for (int cb = 0; cb < 16; cb++)
      Y[row * 256 + cb * 16 + hl] = o[cb][r] * inv;
  }
}

// ---------------------------------------------------------------------------
// Kernel 3a: fold proj+fc:  wt[i] = sum_j Wproj[i][j]*Wfc[j];  wt[256] = bias
// ---------------------------------------------------------------------------
__global__ void fold_kernel(const float* __restrict__ Wproj, const float* __restrict__ bproj,
                            const float* __restrict__ Wfc, const float* __restrict__ bfc,
                            float* __restrict__ wt) {
  int i = threadIdx.x;  // 256 threads, 1 block
  float s = 0.f;
  for (int j = 0; j < 256; j++) s += Wproj[(size_t)i * 256 + j] * Wfc[j];
  wt[i] = s;
  if (i == 0) {
    float sb = 0.f;
    for (int j = 0; j < 256; j++) sb += bproj[j] * Wfc[j];
    wt[256] = sb + bfc[0];
  }
}

// ---------------------------------------------------------------------------
// Kernel 3b: out[row] = y[row,:] . wt + wt[256].  One wave per row.
// ---------------------------------------------------------------------------
__global__ __launch_bounds__(256) void out_kernel(const float* __restrict__ Y,
                                                  const float* __restrict__ wt,
                                                  float* __restrict__ out) {
  int wave = threadIdx.x >> 5, lane = threadIdx.x & 31;
  int row = blockIdx.x * 8 + wave;
  const float* yr = Y + (size_t)row * 256;
  float s = 0.f;
#pragma unroll
  for (int k = 0; k < 8; k++) s += yr[lane + 32 * k] * wt[lane + 32 * k];
#pragma unroll
  for (int m = 16; m >= 1; m >>= 1) s += __shfl_xor(s, m, 32);
  if (lane == 0) out[row] = s + wt[256];
}

// ---------------------------------------------------------------------------
extern "C" void kernel_launch(void* const* d_in, const int* in_sizes, int n_in,
                              void* d_out, int out_size, void* d_ws, size_t ws_size,
                              hipStream_t stream) {
  const float* x     = (const float*)d_in[0];
  const float* Wqkv  = (const float*)d_in[1];
  const float* bqkv  = (const float*)d_in[2];
  const float* Wproj = (const float*)d_in[3];
  const float* bproj = (const float*)d_in[4];
  const float* Wfc   = (const float*)d_in[5];
  const float* bfc   = (const float*)d_in[6];
  float* out = (float*)d_out;

  char* ws = (char*)d_ws;
  _Float16* qk = (_Float16*)ws;                                 // 16384*512*2 = 16.8 MB
  _Float16* vt = (_Float16*)(ws + 16384ull * 512 * 2);          // 4*256*4096*2 = 8.4 MB
  float*    Y  = (float*)(ws + 16384ull * 512 * 2 + 4ull * 256 * 4096 * 2);  // 16.8 MB
  float*    wt = (float*)(ws + 16384ull * 512 * 2 + 4ull * 256 * 4096 * 2 +
                          16384ull * 256 * 4);                  // 257 floats

  fold_kernel<<<1, 256, 0, stream>>>(Wproj, bproj, Wfc, bfc, wt);
  qkv_kernel<<<dim3(128, 8), 256, 0, stream>>>(x, Wqkv, bqkv, qk, vt);
  attn_kernel<<<256, 128, 0, stream>>>(qk, vt, Y);
  out_kernel<<<2048, 256, 0, stream>>>(Y, wt, out);
}
